// Model_64518998721197
// MI455X (gfx1250) — matrix operand order repaired and checked by reading, not compiled, not run on hardware
//
#include <hip/hip_runtime.h>
#include <hip/hip_bf16.h>
#include <math.h>

#define NN   4096
#define D0C  256
#define D1C  128
#define HID  64
#define OUTD 16
#define RREL 3
#define MREL 2
#define KREL 2
#define PROJ 128

typedef __bf16 v8bf  __attribute__((ext_vector_type(8)));
typedef __bf16 v16bf __attribute__((ext_vector_type(16)));
typedef float  v8f   __attribute__((ext_vector_type(8)));

__device__ __forceinline__ unsigned short f2bf_u16(float f) {
  __bf16 h = (__bf16)f;                       // v_cvt_pk_bf16_f32 path
  return __builtin_bit_cast(unsigned short, h);
}

// pack 8 fp32 -> 8 bf16 (backend pairs these into v_cvt_pk_bf16_f32)
__device__ __forceinline__ v8bf cvt8(float4 f0, float4 f1) {
  v8bf r;
  r[0] = (__bf16)f0.x; r[1] = (__bf16)f0.y; r[2] = (__bf16)f0.z; r[3] = (__bf16)f0.w;
  r[4] = (__bf16)f1.x; r[5] = (__bf16)f1.y; r[6] = (__bf16)f1.z; r[7] = (__bf16)f1.w;
  return r;
}

__device__ __forceinline__ float wave_sum(float v) {
#pragma unroll
  for (int off = 16; off > 0; off >>= 1) v += __shfl_xor(v, off, 32);
  return v;
}

// ---------------------------------------------------------------------------
// h^T = bf16(tanh(x @ W + b)), stored transposed [HID][NN] for WMMA B loads
// ---------------------------------------------------------------------------
__global__ __launch_bounds__(256) void hgnn_feat(
    const float* __restrict__ x, const float* __restrict__ W,
    const float* __restrict__ b, unsigned short* __restrict__ hT, int D) {
  const int n = blockIdx.x * 256 + threadIdx.x;
  const int d = blockIdx.y;
  const float* xr = x + (size_t)n * D;
  float acc = b[d];
#pragma unroll 4
  for (int k = 0; k < D; ++k) acc += xr[k] * W[(size_t)k * HID + d];
  hT[(size_t)d * NN + n] = f2bf_u16(tanhf(acc));
}

// ---------------------------------------------------------------------------
// out[rel] = (tanh?) (adj[rel] @ h)    adj fp32 [NN,NN], h given as hT bf16
// block = 256 thr = 8 waves, tile = 128 rows x 64 cols, K-chunk = 32
// double-buffered LDS software pipeline: load(k+1) | wmma(k) | store(k+1) | bar
// ---------------------------------------------------------------------------
__global__ __launch_bounds__(256) void hgnn_agg_gemm(
    const float* __restrict__ adj, const unsigned short* __restrict__ hT,
    float* __restrict__ outp, int apply_tanh) {
  __shared__ unsigned short lds_a[2][128][40];  // bf16 adjacency tiles (+pad)
  __shared__ unsigned short lds_b[2][64][40];   // bf16 hT tiles

  const int tid  = threadIdx.x;
  const int wave = tid >> 5;
  const int lane = tid & 31;
  const int rel  = blockIdx.y;
  const int row0 = blockIdx.x * 128;

  const float* A = adj + (size_t)rel * NN * NN;
  float*       C = outp + (size_t)rel * NN * HID;

  v8f acc0 = {}, acc1 = {}, acc2 = {}, acc3 = {};

  const int ar_st = tid >> 1;          // staging: A row 0..127
  const int ac_st = (tid & 1) * 16;    // staging: A col 0/16
  const int br_st = tid >> 2;          // staging: B row 0..63
  const int bc_st = (tid & 3) * 8;     // staging: B col 0/8/16/24

  const float*          aptr = A  + (size_t)(row0 + ar_st) * NN + ac_st;
  const unsigned short* bptr = hT + (size_t)br_st * NN + bc_st;

  const int ar = wave * 16 + (lane & 15);   // A fragment row within tile
  const int ak = (lane < 16) ? 0 : 8;       // A: K octets {ak..ak+7, ak+16..ak+23}
  const int bk = (lane < 16) ? 0 : 8;       // B: K octets {bk..bk+7, bk+16..bk+23}

  float4 f0, f1, f2, f3;
  uint4  bv;

  // prologue: stage K-chunk 0 into buffer 0
  f0 = reinterpret_cast<const float4*>(aptr)[0];
  f1 = reinterpret_cast<const float4*>(aptr)[1];
  f2 = reinterpret_cast<const float4*>(aptr)[2];
  f3 = reinterpret_cast<const float4*>(aptr)[3];
  bv = *reinterpret_cast<const uint4*>(bptr);
  *reinterpret_cast<v8bf*>(&lds_a[0][ar_st][ac_st])     = cvt8(f0, f1);
  *reinterpret_cast<v8bf*>(&lds_a[0][ar_st][ac_st + 8]) = cvt8(f2, f3);
  *reinterpret_cast<uint4*>(&lds_b[0][br_st][bc_st])    = bv;
  __syncthreads();

  for (int kc = 0; kc < NN; kc += 32) {
    const int  buf  = (kc >> 5) & 1;
    const bool more = (kc + 32) < NN;

    if (more) {  // issue next tile's global loads early (overlap with WMMA)
      const float* ap = aptr + kc + 32;
      f0 = reinterpret_cast<const float4*>(ap)[0];
      f1 = reinterpret_cast<const float4*>(ap)[1];
      f2 = reinterpret_cast<const float4*>(ap)[2];
      f3 = reinterpret_cast<const float4*>(ap)[3];
      bv = *reinterpret_cast<const uint4*>(bptr + kc + 32);
      if (kc + 64 < NN)  // pull the tile after that toward the WGP
        __builtin_prefetch(aptr + kc + 64, 0, 1);
    }

    // A fragment (ISA 16-bit A 16x32 layout)
    v8bf alo = *reinterpret_cast<const v8bf*>(&lds_a[buf][ar][ak]);
    v8bf ahi = *reinterpret_cast<const v8bf*>(&lds_a[buf][ar][ak + 16]);
    v16bf af = __builtin_shufflevector(alo, ahi, 0,1,2,3,4,5,6,7,8,9,10,11,12,13,14,15);

#define DO_G(g, accv)                                                                         \
    {                                                                                         \
      v8bf blo = *reinterpret_cast<const v8bf*>(&lds_b[buf][(g) * 16 + (lane & 15)][bk]);     \
      v8bf bhi = *reinterpret_cast<const v8bf*>(&lds_b[buf][(g) * 16 + (lane & 15)][bk + 16]); \
      v16bf bf = __builtin_shufflevector(blo, bhi,                                            \
                                         0,1,2,3,4,5,6,7,8,9,10,11,12,13,14,15);              \
      accv = __builtin_amdgcn_wmma_f32_16x16x32_bf16(false, af, false, bf,                    \
                                                     (short)0, accv, false, false);           \
    }
    DO_G(0, acc0) DO_G(1, acc1) DO_G(2, acc2) DO_G(3, acc3)
#undef DO_G

    if (more) {  // convert + store next tile into the other buffer
      *reinterpret_cast<v8bf*>(&lds_a[buf ^ 1][ar_st][ac_st])     = cvt8(f0, f1);
      *reinterpret_cast<v8bf*>(&lds_a[buf ^ 1][ar_st][ac_st + 8]) = cvt8(f2, f3);
      *reinterpret_cast<uint4*>(&lds_b[buf ^ 1][br_st][bc_st])    = bv;
    }
    __syncthreads();
  }

  // C layout: VGPR j -> M=j (lanes 0-15) / M=j+8 (lanes 16-31), N = lane&15
  const int crow_base = row0 + wave * 16 + ((lane < 16) ? 0 : 8);
  const int ccol      = lane & 15;
#pragma unroll
  for (int j = 0; j < 8; ++j) {
    float v0 = acc0[j], v1 = acc1[j], v2 = acc2[j], v3 = acc3[j];
    if (apply_tanh) { v0 = tanhf(v0); v1 = tanhf(v1); v2 = tanhf(v2); v3 = tanhf(v3); }
    float* cp = C + (size_t)(crow_base + j) * HID + ccol;
    cp[0] = v0; cp[16] = v1; cp[32] = v2; cp[48] = v3;
  }
}

// ---------------------------------------------------------------------------
// semantic attention scores: s[r] += mean_n tanh(hr[r,n,:]@Wa+ba)@qa
// ---------------------------------------------------------------------------
__global__ __launch_bounds__(256) void hgnn_meta_score(
    const float* __restrict__ hr, const float* __restrict__ Wa,
    const float* __restrict__ ba, const float* __restrict__ qa,
    float* __restrict__ sbuf) {
  __shared__ float lh[8][64];
  const int tid = threadIdx.x, wave = tid >> 5, lane = tid & 31;
  const int r = blockIdx.y;
  const int n = blockIdx.x * 8 + wave;
  const float* row = hr + ((size_t)r * NN + n) * HID;
  lh[wave][lane] = row[lane];
  lh[wave][lane + 32] = row[lane + 32];
  __syncthreads();
  float part = 0.f;
#pragma unroll
  for (int pp = 0; pp < 4; ++pp) {
    const int p = lane * 4 + pp;
    float acc = ba[p];
    for (int d = 0; d < HID; ++d) acc += lh[wave][d] * Wa[(size_t)d * PROJ + p];
    part += tanhf(acc) * qa[p];
  }
  part = wave_sum(part);
  if (lane == 0) atomicAdd(&sbuf[r], part * (1.0f / NN));
}

__global__ void hgnn_softmax_r(float* sbuf) {
  if (threadIdx.x == 0) {
    float m = fmaxf(sbuf[0], fmaxf(sbuf[1], sbuf[2]));
    float e0 = expf(sbuf[0] - m), e1 = expf(sbuf[1] - m), e2 = expf(sbuf[2] - m);
    float inv = 1.f / (e0 + e1 + e2);
    sbuf[4] = e0 * inv; sbuf[5] = e1 * inv; sbuf[6] = e2 * inv;
  }
}

__global__ __launch_bounds__(256) void hgnn_relagg(
    const float* __restrict__ hr, const float* __restrict__ sbuf,
    float* __restrict__ ragg) {
  const size_t idx = (size_t)blockIdx.x * 256 + threadIdx.x;
  const size_t S = (size_t)NN * HID;
  ragg[idx] = sbuf[4] * hr[idx] + sbuf[5] * hr[S + idx] + sbuf[6] * hr[2 * S + idx];
}

// ---------------------------------------------------------------------------
// schema: per-node softmax over K=2 of zk@ws; write bf16 transposed combine
// ---------------------------------------------------------------------------
__global__ __launch_bounds__(256) void hgnn_schema(
    const float* __restrict__ zk, const float* __restrict__ wsv,
    unsigned short* __restrict__ icT) {
  const int tid = threadIdx.x, wave = tid >> 5, lane = tid & 31;
  const int n = blockIdx.x * 8 + wave;
  const float* z0 = zk + (size_t)n * HID;
  const float* z1 = zk + (size_t)NN * HID + (size_t)n * HID;
  const float a0 = z0[lane], a1 = z0[lane + 32];
  const float b0 = z1[lane], b1 = z1[lane + 32];
  const float w0 = wsv[lane], w1 = wsv[lane + 32];
  const float s0 = wave_sum(a0 * w0 + a1 * w1);
  const float s1 = wave_sum(b0 * w0 + b1 * w1);
  const float m = fmaxf(s0, s1);
  float e0 = expf(s0 - m), e1 = expf(s1 - m);
  const float inv = 1.f / (e0 + e1);
  e0 *= inv; e1 *= inv;
  icT[(size_t)lane * NN + n]        = f2bf_u16(e0 * a0 + e1 * b0);
  icT[(size_t)(lane + 32) * NN + n] = f2bf_u16(e0 * a1 + e1 * b1);
}

// ---------------------------------------------------------------------------
// same-type relations: wm, softmax over M=2, item_item combine
// ---------------------------------------------------------------------------
__global__ __launch_bounds__(256) void hgnn_same(
    const float* __restrict__ ym, const float* __restrict__ Wc1,
    const float* __restrict__ bc1, const float* __restrict__ wc2,
    float* __restrict__ iitem) {
  __shared__ float ly[8][2][64];
  const int tid = threadIdx.x, wave = tid >> 5, lane = tid & 31;
  const int n = blockIdx.x * 8 + wave;
  const size_t S = (size_t)NN * HID;
#pragma unroll
  for (int m = 0; m < 2; ++m) {
    const float* yr = ym + m * S + (size_t)n * HID;
    ly[wave][m][lane] = yr[lane];
    ly[wave][m][lane + 32] = yr[lane + 32];
  }
  __syncthreads();
  float w[2];
#pragma unroll
  for (int m = 0; m < 2; ++m) {
    float part = 0.f;
#pragma unroll
    for (int pp = 0; pp < 4; ++pp) {
      const int p = lane * 4 + pp;
      float acc = bc1[p];
      for (int d = 0; d < HID; ++d) acc += ly[wave][m][d] * Wc1[(size_t)d * PROJ + p];
      part += tanhf(acc) * wc2[p];
    }
    w[m] = wave_sum(part);
  }
  const float mm = fmaxf(w[0], w[1]);
  float e0 = expf(w[0] - mm), e1 = expf(w[1] - mm);
  const float inv = 1.f / (e0 + e1);
  e0 *= inv; e1 *= inv;
  float* outr = iitem + (size_t)n * HID;
  outr[lane]      = e0 * ly[wave][0][lane]      + e1 * ly[wave][1][lane];
  outr[lane + 32] = e0 * ly[wave][0][lane + 32] + e1 * ly[wave][1][lane + 32];
}

// ---------------------------------------------------------------------------
// final: AggAttention over {relation_agg, item_item}, H, out = H@Wp+bp
// ---------------------------------------------------------------------------
__global__ __launch_bounds__(256) void hgnn_final(
    const float* __restrict__ ragg, const float* __restrict__ iitem,
    const float* __restrict__ Wt1, const float* __restrict__ bt1,
    const float* __restrict__ wt2, const float* __restrict__ Wp,
    const float* __restrict__ bp, float* __restrict__ out1,
    float* __restrict__ outH) {
  __shared__ float lh[8][2][64];
  __shared__ float lH[8][64];
  const int tid = threadIdx.x, wave = tid >> 5, lane = tid & 31;
  const int n = blockIdx.x * 8 + wave;
  {
    const float* r0 = ragg + (size_t)n * HID;
    const float* r1 = iitem + (size_t)n * HID;
    lh[wave][0][lane] = r0[lane]; lh[wave][0][lane + 32] = r0[lane + 32];
    lh[wave][1][lane] = r1[lane]; lh[wave][1][lane + 32] = r1[lane + 32];
  }
  __syncthreads();
  float w[2];
#pragma unroll
  for (int t = 0; t < 2; ++t) {
    float part = 0.f;
#pragma unroll
    for (int pp = 0; pp < 4; ++pp) {
      const int p = lane * 4 + pp;
      float acc = bt1[p];
      for (int d = 0; d < HID; ++d) acc += lh[wave][t][d] * Wt1[(size_t)d * PROJ + p];
      part += tanhf(acc) * wt2[p];
    }
    w[t] = wave_sum(part);
  }
  const float mm = fmaxf(w[0], w[1]);
  float e0 = expf(w[0] - mm), e1 = expf(w[1] - mm);
  const float inv = 1.f / (e0 + e1);
  e0 *= inv; e1 *= inv;
  const float h0v = e0 * lh[wave][0][lane]      + e1 * lh[wave][1][lane];
  const float h1v = e0 * lh[wave][0][lane + 32] + e1 * lh[wave][1][lane + 32];
  lH[wave][lane] = h0v; lH[wave][lane + 32] = h1v;
  outH[(size_t)n * HID + lane] = h0v;
  outH[(size_t)n * HID + lane + 32] = h1v;
  __syncthreads();
  if (lane < OUTD) {
    float acc = bp[lane];
    for (int d = 0; d < HID; ++d) acc += lH[wave][d] * Wp[(size_t)d * OUTD + lane];
    out1[(size_t)n * OUTD + lane] = acc;
  }
}

__global__ void hgnn_zero(float* p, int cnt) {
  const int i = blockIdx.x * 256 + threadIdx.x;
  if (i < cnt) p[i] = 0.f;
}

// ---------------------------------------------------------------------------
extern "C" void kernel_launch(void* const* d_in, const int* in_sizes, int n_in,
                              void* d_out, int out_size, void* d_ws, size_t ws_size,
                              hipStream_t stream) {
  (void)in_sizes; (void)n_in; (void)out_size; (void)ws_size;

  const float* features0  = (const float*)d_in[0];
  const float* features1  = (const float*)d_in[1];
  const float* adj_same   = (const float*)d_in[2];
  const float* adj_meta   = (const float*)d_in[3];
  const float* adj_schema = (const float*)d_in[4];
  const float* Wm0 = (const float*)d_in[5];  const float* bm0 = (const float*)d_in[6];
  const float* Wm1 = (const float*)d_in[7];  const float* bm1 = (const float*)d_in[8];
  const float* Wa  = (const float*)d_in[9];  const float* ba  = (const float*)d_in[10];
  const float* qa  = (const float*)d_in[11];
  const float* wsv = (const float*)d_in[12];
  const float* Wc1 = (const float*)d_in[13]; const float* bc1 = (const float*)d_in[14];
  const float* wc2 = (const float*)d_in[15];
  const float* Wt1 = (const float*)d_in[16]; const float* bt1 = (const float*)d_in[17];
  const float* wt2 = (const float*)d_in[18];
  const float* Wp  = (const float*)d_in[19]; const float* bp  = (const float*)d_in[20];

  char* ws = (char*)d_ws;
  unsigned short* h0T = (unsigned short*)(ws);              // bf16 [64][4096]
  unsigned short* h1T = (unsigned short*)(ws + 524288);     // bf16 [64][4096]
  unsigned short* icT = (unsigned short*)(ws + 1048576);    // bf16 [64][4096]
  float* hr    = (float*)(ws + 1572864);                    // [3][4096][64]
  float* zk    = (float*)(ws + 4718592);                    // [2][4096][64]
  float* ym    = (float*)(ws + 6815744);                    // [2][4096][64]
  float* ragg  = (float*)(ws + 8912896);                    // [4096][64]
  float* iitem = (float*)(ws + 9961472);                    // [4096][64]
  float* sbuf  = (float*)(ws + 11010048);                   // scores + betas

  float* out1 = (float*)d_out;
  float* outH = out1 + (size_t)NN * OUTD;

  hgnn_zero<<<1, 256, 0, stream>>>(sbuf, 8);

  dim3 fgrid(NN / 256, HID);
  hgnn_feat<<<fgrid, 256, 0, stream>>>(features0, Wm0, bm0, h0T, D0C);
  hgnn_feat<<<fgrid, 256, 0, stream>>>(features1, Wm1, bm1, h1T, D1C);

  hgnn_agg_gemm<<<dim3(NN / 128, RREL), 256, 0, stream>>>(adj_meta, h0T, hr, 1);
  hgnn_meta_score<<<dim3(NN / 8, RREL), 256, 0, stream>>>(hr, Wa, ba, qa, sbuf);
  hgnn_softmax_r<<<1, 32, 0, stream>>>(sbuf);
  hgnn_relagg<<<(NN * HID) / 256, 256, 0, stream>>>(hr, sbuf, ragg);

  hgnn_agg_gemm<<<dim3(NN / 128, KREL), 256, 0, stream>>>(adj_schema, h1T, zk, 0);
  hgnn_schema<<<NN / 8, 256, 0, stream>>>(zk, wsv, icT);

  hgnn_agg_gemm<<<dim3(NN / 128, MREL), 256, 0, stream>>>(adj_same, icT, ym, 1);
  hgnn_same<<<NN / 8, 256, 0, stream>>>(ym, Wc1, bc1, wc2, iitem);

  hgnn_final<<<NN / 8, 256, 0, stream>>>(ragg, iitem, Wt1, bt1, wt2, Wp, bp, out1, outH);
}
